// Convolutional_19189913879087
// MI455X (gfx1250) — matrix operand order repaired and checked by reading, not compiled, not run on hardware
//
#include <hip/hip_runtime.h>

typedef __attribute__((ext_vector_type(16))) _Float16     v16h;
typedef __attribute__((ext_vector_type(8)))  float        v8f;
typedef __attribute__((ext_vector_type(4)))  unsigned int u32x4;

#define C_IN  64
#define D_OUT 128
#define H_IN  512
#define W_IN  512
#define HO    510
#define WO    510

// tile configuration
#define OX_TILE 64
#define OY_TILE 4
#define ROWS (OY_TILE + 2)          // 6 input rows staged
#define XS   (OX_TILE + 2)          // 66 input cols staged
#define CPAD 72                     // padded channel stride (f16): 144B rows -> 16B aligned, bank-friendly
#define SMEM_HALFS (ROWS * XS * CPAD)
#define SMEM_BYTES (SMEM_HALFS * 2) // 57,024 B < 64 KB

// ---------------------------------------------------------------------------
// Pack kernels [128][64][3][3] f32 -> f16 A-fragments in the exact per-lane
// WMMA 16-bit A-matrix (16x32) layout, indexed [mtile][fidx][lane][16 halves],
// where fidx = (kh*3+kw)*2 + ck and the 32-wide K chunk maps to c = ck*32 + K.
// ---------------------------------------------------------------------------
__global__ void pack_A(const float* __restrict__ k4, _Float16* __restrict__ ap) {
  int idx = blockIdx.x * 256 + threadIdx.x;           // 8*18*32*16 = 73728 halves
  if (idx >= 8 * 18 * 32 * 16) return;
  int e    = idx & 15;
  int t    = idx >> 4;
  int lane = t & 31;
  t >>= 5;
  int f  = t % 18;
  int mt = t / 18;
  int i = e >> 1, h = e & 1, g = lane >> 4, m = lane & 15;
  // ISA 16-bit A-matrix 16x32 per-lane K mapping
  int K  = (i < 4) ? (8 * g + 2 * i + h) : (16 + 8 * g + 2 * (i - 4) + h);
  int ck = f & 1;
  int p  = f >> 1;
  int kh = p / 3, kw = p % 3;
  int c = ck * 32 + K;
  int M = mt * 16 + m;
  ap[idx] = (_Float16)k4[((M * C_IN + c) * 3 + kh) * 3 + kw];
}

// ---------------------------------------------------------------------------
// Implicit-GEMM conv: 9 shifted pointwise GEMMs, f16 operands, f32 accum.
// Workgroup: 256 threads (8 waves) = 4 channel-groups x 2 pixel-groups.
// Each wave: 2 M-tiles (32 ch) x 2 oxt x 2 oyr accumulators (8x v8f).
// Per k-step: 2 A frags + 4 B frags -> 8 WMMAs (1 ds_b128 per WMMA).
// ---------------------------------------------------------------------------
__global__ void __launch_bounds__(256)
conv_wmma(const float* __restrict__ in,
          const v16h*  __restrict__ Ap,
          const float* __restrict__ bias,
          float*       __restrict__ out) {
  extern __shared__ _Float16 smem[];
  const int ox0 = blockIdx.x * OX_TILE;
  const int oy0 = blockIdx.y * OY_TILE;
  const int tid = threadIdx.x;

  // Stage input patch [row][x][c] (channel-fastest, padded) as f16.
  const int total = C_IN * ROWS * XS;                 // 25344 = 99 * 256 exactly
  for (int idx = tid; idx < total; idx += 256) {
    int x   = idx % XS;
    int t   = idx / XS;
    int row = t % ROWS;
    int c   = t / ROWS;
    int gx = ox0 + x, gy = oy0 + row;
    float v = 0.0f;
    if (gx < W_IN && gy < H_IN) v = in[(c * H_IN + gy) * W_IN + gx];
    smem[(row * XS + x) * CPAD + c] = (_Float16)v;
  }
  __syncthreads();

  const int lane = tid & 31;
  const int wave = tid >> 5;
  const int mg   = wave >> 1;     // channel group: m-tiles {2mg, 2mg+1}
  const int pg   = wave & 1;      // pixel group: oxt {2pg, 2pg+1}
  const int g    = lane >> 4;     // half-wave group
  const int n    = lane & 15;     // N (pixel) index within tile

  union BFrag { u32x4 q[2]; v16h h; };

  for (int oy2 = 0; oy2 < 2; ++oy2) {   // two output-row pairs
    v8f acc[2][2][2] = {};              // [mi][yi][xi]

#pragma unroll
    for (int kh = 0; kh < 3; ++kh) {
#pragma unroll
      for (int kw = 0; kw < 3; ++kw) {
#pragma unroll
        for (int ck = 0; ck < 2; ++ck) {
          const int f = (kh * 3 + kw) * 2 + ck;
          // Two A fragments (32B/lane each), coalesced & L2-hot.
          v16h a0 = Ap[((mg * 2 + 0) * 18 + f) * 32 + lane];
          v16h a1 = Ap[((mg * 2 + 1) * 18 + f) * 32 + lane];
#pragma unroll
          for (int yi = 0; yi < 2; ++yi) {
            const int row = oy2 * 2 + yi + kh;
#pragma unroll
            for (int xi = 0; xi < 2; ++xi) {
              const int xb = (pg * 2 + xi) * 16 + n + kw;
              const u32x4* bp =
                  (const u32x4*)&smem[(row * XS + xb) * CPAD + ck * 32 + g * 8];
              BFrag bf;
              bf.q[0] = bp[0];
              bf.q[1] = bp[2];
              acc[0][yi][xi] = __builtin_amdgcn_wmma_f32_16x16x32_f16(
                  false, a0, false, bf.h, (short)0, acc[0][yi][xi], false, false);
              acc[1][yi][xi] = __builtin_amdgcn_wmma_f32_16x16x32_f16(
                  false, a1, false, bf.h, (short)0, acc[1][yi][xi], false, false);
            }
          }
        }
      }
    }

    // Epilogue: bias add + f32 store, non-temporal (touch-once streams).
    // C/D layout: VGPR j -> M = j + 8g, N = n.
#pragma unroll
    for (int yi = 0; yi < 2; ++yi) {
      const int oy = oy0 + oy2 * 2 + yi;
      if (oy < HO) {
#pragma unroll
        for (int xi = 0; xi < 2; ++xi) {
          const int ox = ox0 + (pg * 2 + xi) * 16 + n;
          if (ox < WO) {
#pragma unroll
            for (int mi = 0; mi < 2; ++mi) {
#pragma unroll
              for (int j = 0; j < 8; ++j) {
                const int d = (mg * 2 + mi) * 16 + g * 8 + j;
                const int o = (d * HO + oy) * WO + ox;
                float b = __builtin_nontemporal_load(&bias[o]);
                __builtin_nontemporal_store(acc[mi][yi][xi][j] + b, &out[o]);
              }
            }
          }
        }
      }
    }
  }
}

extern "C" void kernel_launch(void* const* d_in, const int* in_sizes, int n_in,
                              void* d_out, int out_size, void* d_ws, size_t ws_size,
                              hipStream_t stream) {
  const float* input   = (const float*)d_in[0];   // [64,512,512] f32
  const float* kernels = (const float*)d_in[1];   // [128,64,3,3] f32
  const float* biases  = (const float*)d_in[2];   // [128,510,510] f32
  float*       out     = (float*)d_out;           // [128,510,510] f32
  _Float16*    apack   = (_Float16*)d_ws;         // 73728 f16 = 144 KB scratch

  // Pack kernels into WMMA A-fragment layout (f32 -> f16).
  pack_A<<<(8 * 18 * 32 * 16 + 255) / 256, 256, 0, stream>>>(kernels, apack);

  // Implicit-GEMM conv: 8 ox-tiles x 128 oy-tiles.
  dim3 grid((WO + OX_TILE - 1) / OX_TILE, (HO + OY_TILE - 1) / OY_TILE);
  conv_wmma<<<grid, 256, SMEM_BYTES, stream>>>(input, (const v16h*)apack, biases, out);
}
